// GroupedMOE_69234872811783
// MI455X (gfx1250) — matrix operand order repaired and checked by reading, not compiled, not run on hardware
//
#include <hip/hip_runtime.h>

// Problem constants (match reference)
#define TT 4096   // B*S tokens
#define DD 1024   // model dim
#define FF 1024   // ffn dim
#define EE 8      // experts
// GEMM tiling
#define BM 128
#define BN 128
#define BK 32
#define LDH 48    // LDS row stride in halfs (padded, 96B, multiple of 16B)

typedef __attribute__((ext_vector_type(16))) __bf16 v16bf;
typedef __attribute__((ext_vector_type(8)))  __bf16 v8bf;
typedef __attribute__((ext_vector_type(8)))  float  v8f;

// ---- WMMA fragment loaders (CDNA5 ISA 7.12.2 layouts, wave32) ----
// A (16x32 bf16): lanes 0-15 hold row M=lane, K={0..7,16..23}; lanes 16-31 K={8..15,24..31}
__device__ __forceinline__ v16bf ld_fragA(const __bf16* row, int half) {
    union { v16bf v; v8bf h[2]; } u;
    u.h[0] = *reinterpret_cast<const v8bf*>(row + half * 8);
    u.h[1] = *reinterpret_cast<const v8bf*>(row + 16 + half * 8);
    return u.v;
}
// B (32x16 bf16): lanes 0-15 hold col N=lane, K={0..7,16..23}; lanes 16-31 K={8..15,24..31}
__device__ __forceinline__ v16bf ld_fragB(const __bf16* row, int half) {
    union { v16bf v; v8bf h[2]; } u;
    u.h[0] = *reinterpret_cast<const v8bf*>(row + half * 8);
    u.h[1] = *reinterpret_cast<const v8bf*>(row + 16 + half * 8);
    return u.v;
}

__device__ __forceinline__ void cvt4(__bf16* dst, float4 v) {
    dst[0] = (__bf16)v.x; dst[1] = (__bf16)v.y;
    dst[2] = (__bf16)v.z; dst[3] = (__bf16)v.w;
}

// ---------------- Gating: logits -> softmax -> top2 -> renorm dense gate ----------------
__global__ __launch_bounds__(256) void moe_gate_kernel(
    const float* __restrict__ x, const float* __restrict__ gw,
    float* __restrict__ gate_full) {
    __shared__ float gws[EE * DD];   // 32 KB
    for (int i = threadIdx.x; i < EE * DD; i += 256) gws[i] = gw[i];
    __syncthreads();

    const int t = blockIdx.x * 256 + threadIdx.x;
    const float* xr = x + (size_t)t * DD;
    float acc[EE];
#pragma unroll
    for (int e = 0; e < EE; ++e) acc[e] = 0.f;
    for (int d = 0; d < DD; d += 4) {
        float4 xv = *reinterpret_cast<const float4*>(xr + d);
#pragma unroll
        for (int e = 0; e < EE; ++e) {
            const float* g = &gws[e * DD + d];
            acc[e] += xv.x * g[0] + xv.y * g[1] + xv.z * g[2] + xv.w * g[3];
        }
    }
    float m = acc[0];
#pragma unroll
    for (int e = 1; e < EE; ++e) m = fmaxf(m, acc[e]);
    float p[EE], s = 0.f;
#pragma unroll
    for (int e = 0; e < EE; ++e) { p[e] = __expf(acc[e] - m); s += p[e]; }
    float inv = 1.f / s;
#pragma unroll
    for (int e = 0; e < EE; ++e) p[e] *= inv;
    int i1 = 0;
#pragma unroll
    for (int e = 1; e < EE; ++e) if (p[e] > p[i1]) i1 = e;
    int i2 = (i1 == 0) ? 1 : 0;
#pragma unroll
    for (int e = 0; e < EE; ++e) if (e != i2 && e != i1 && p[e] > p[i2]) i2 = e;
    float inv2 = 1.f / (p[i1] + p[i2]);
#pragma unroll
    for (int e = 0; e < EE; ++e) gate_full[t * EE + e] = 0.f;
    gate_full[t * EE + i1] = p[i1] * inv2;
    gate_full[t * EE + i2] = p[i2] * inv2;
}

// ---------------- GEMM1: H = silu(X @ W1e^T) * (X @ W3e^T), bf16 out ----------------
// grid = (TT/BM, FF/BN), block = 256 = 8 waves in 2(M) x 4(N), wave tile 64x32.
// Register-staged double buffering: next K-tile loads overlap WMMA on current tile.
__global__ __launch_bounds__(256) void moe_gemm1_kernel(
    const float* __restrict__ x, const float* __restrict__ w1e,
    const float* __restrict__ w3e, __bf16* __restrict__ H) {
    __shared__ __align__(16) __bf16 Xs[BM * LDH];
    __shared__ __align__(16) __bf16 W1s[BN * LDH];
    __shared__ __align__(16) __bf16 W3s[BN * LDH];

    const int tid = threadIdx.x;
    const int lane = tid & 31, wave = tid >> 5;
    const int half = lane >> 4, r = lane & 15;
    const int waveM = (wave >> 2) * 64, waveN = (wave & 3) * 32;
    const int tileM = blockIdx.x * BM, tileN = blockIdx.y * BN;
    // staging map: q = tid + u*256 -> row = q>>3 (0..127), c4 = (q&7)*4
    const int srow = tid >> 3, sc4 = (tid & 7) * 4;

    v8f acc1[4][2] = {};
    v8f acc3[4][2] = {};
    float4 xr_[4], w1r_[4], w3r_[4];

    // preload first K-tile into registers
#pragma unroll
    for (int u = 0; u < 4; ++u) {
        const int row = srow + u * 32;
        xr_[u]  = *reinterpret_cast<const float4*>(x   + (size_t)(tileM + row) * DD + sc4);
        w1r_[u] = *reinterpret_cast<const float4*>(w1e + (size_t)(tileN + row) * DD + sc4);
        w3r_[u] = *reinterpret_cast<const float4*>(w3e + (size_t)(tileN + row) * DD + sc4);
    }

    for (int kb = 0; kb < DD; kb += BK) {
        // commit staged registers to LDS (fp32 -> bf16)
#pragma unroll
        for (int u = 0; u < 4; ++u) {
            const int row = srow + u * 32;
            cvt4(&Xs[row * LDH + sc4], xr_[u]);
            cvt4(&W1s[row * LDH + sc4], w1r_[u]);
            cvt4(&W3s[row * LDH + sc4], w3r_[u]);
        }
        __syncthreads();

        // issue next K-tile global loads (latency overlapped with WMMAs below)
        if (kb + BK < DD) {
            const int kn = kb + BK;
#pragma unroll
            for (int u = 0; u < 4; ++u) {
                const int row = srow + u * 32;
                xr_[u]  = *reinterpret_cast<const float4*>(x   + (size_t)(tileM + row) * DD + kn + sc4);
                w1r_[u] = *reinterpret_cast<const float4*>(w1e + (size_t)(tileN + row) * DD + kn + sc4);
                w3r_[u] = *reinterpret_cast<const float4*>(w3e + (size_t)(tileN + row) * DD + kn + sc4);
            }
        }

        v16bf b1fr[2], b3fr[2];
#pragma unroll
        for (int s = 0; s < 2; ++s) {
            b1fr[s] = ld_fragB(&W1s[(waveN + s * 16 + r) * LDH], half);
            b3fr[s] = ld_fragB(&W3s[(waveN + s * 16 + r) * LDH], half);
        }
#pragma unroll
        for (int i = 0; i < 4; ++i) {
            v16bf afr = ld_fragA(&Xs[(waveM + i * 16 + r) * LDH], half);
#pragma unroll
            for (int j = 0; j < 2; ++j) {
                acc1[i][j] = __builtin_amdgcn_wmma_f32_16x16x32_bf16(
                    false, afr, false, b1fr[j], (short)0, acc1[i][j], false, false);
                acc3[i][j] = __builtin_amdgcn_wmma_f32_16x16x32_bf16(
                    false, afr, false, b3fr[j], (short)0, acc3[i][j], false, false);
            }
        }
        __syncthreads();
    }

    // epilogue: h = silu(h1) * h3  (C layout: VGPR k -> M = k + 8*half, N = r)
#pragma unroll
    for (int i = 0; i < 4; ++i)
#pragma unroll
        for (int j = 0; j < 2; ++j)
#pragma unroll
            for (int k = 0; k < 8; ++k) {
                float a = acc1[i][j][k];
                float b = acc3[i][j][k];
                float hv = (a / (1.f + __expf(-a))) * b;
                const int row = tileM + waveM + i * 16 + half * 8 + k;
                const int col = tileN + waveN + j * 16 + r;
                H[(size_t)row * FF + col] = (__bf16)hv;
            }
}

// ---------------- GEMM2: out (+)= gate[:,e] * (H @ W2e^T) ----------------
// grid = (TT/BM, DD/BN), block = 256, wave tile 64x32, double-buffered staging.
__global__ __launch_bounds__(256) void moe_gemm2_kernel(
    const __bf16* __restrict__ H, const float* __restrict__ w2e,
    const float* __restrict__ gate_full, float* __restrict__ out,
    int e, int accum) {
    __shared__ __align__(16) __bf16 As[BM * LDH];
    __shared__ __align__(16) __bf16 Bs[BN * LDH];

    const int tid = threadIdx.x;
    const int lane = tid & 31, wave = tid >> 5;
    const int half = lane >> 4, r = lane & 15;
    const int waveM = (wave >> 2) * 64, waveN = (wave & 3) * 32;
    const int tileM = blockIdx.x * BM, tileN = blockIdx.y * BN;
    // A staging (bf16 copy): q = tid + u*256 -> row = q>>2, c8 = (q&3)*8
    const int arow = tid >> 2, ac8 = (tid & 3) * 8;
    // B staging (fp32 cvt): row = q>>3, c4 = (q&7)*4
    const int brow = tid >> 3, bc4 = (tid & 7) * 4;

    v8f acc[4][2] = {};
    uint4 ar_[2];
    float4 br_[4];

#pragma unroll
    for (int u = 0; u < 2; ++u)
        ar_[u] = *reinterpret_cast<const uint4*>(H + (size_t)(tileM + arow + u * 64) * FF + ac8);
#pragma unroll
    for (int u = 0; u < 4; ++u)
        br_[u] = *reinterpret_cast<const float4*>(w2e + (size_t)(tileN + brow + u * 32) * FF + bc4);

    for (int kb = 0; kb < FF; kb += BK) {
#pragma unroll
        for (int u = 0; u < 2; ++u)
            *reinterpret_cast<uint4*>(&As[(arow + u * 64) * LDH + ac8]) = ar_[u];
#pragma unroll
        for (int u = 0; u < 4; ++u)
            cvt4(&Bs[(brow + u * 32) * LDH + bc4], br_[u]);
        __syncthreads();

        if (kb + BK < FF) {
            const int kn = kb + BK;
#pragma unroll
            for (int u = 0; u < 2; ++u)
                ar_[u] = *reinterpret_cast<const uint4*>(H + (size_t)(tileM + arow + u * 64) * FF + kn + ac8);
#pragma unroll
            for (int u = 0; u < 4; ++u)
                br_[u] = *reinterpret_cast<const float4*>(w2e + (size_t)(tileN + brow + u * 32) * FF + kn + bc4);
        }

        v16bf bfr[2];
#pragma unroll
        for (int s = 0; s < 2; ++s)
            bfr[s] = ld_fragB(&Bs[(waveN + s * 16 + r) * LDH], half);
#pragma unroll
        for (int i = 0; i < 4; ++i) {
            v16bf afr = ld_fragA(&As[(waveM + i * 16 + r) * LDH], half);
#pragma unroll
            for (int j = 0; j < 2; ++j)
                acc[i][j] = __builtin_amdgcn_wmma_f32_16x16x32_bf16(
                    false, afr, false, bfr[j], (short)0, acc[i][j], false, false);
        }
        __syncthreads();
    }

#pragma unroll
    for (int i = 0; i < 4; ++i)
#pragma unroll
        for (int k = 0; k < 8; ++k) {
            const int row = tileM + waveM + i * 16 + half * 8 + k;
            const float g = gate_full[row * EE + e];
#pragma unroll
            for (int j = 0; j < 2; ++j) {
                const int col = tileN + waveN + j * 16 + r;
                float val = g * acc[i][j][k];
                const size_t o = (size_t)row * DD + col;
                if (accum) val += out[o];
                out[o] = val;
            }
        }
}

extern "C" void kernel_launch(void* const* d_in, const int* in_sizes, int n_in,
                              void* d_out, int out_size, void* d_ws, size_t ws_size,
                              hipStream_t stream) {
    const float* x      = (const float*)d_in[0];   // [B,S,D] = [T,D]
    const float* gate_w = (const float*)d_in[1];   // [E,D]
    const float* w1     = (const float*)d_in[2];   // [E,F,D]
    const float* w3     = (const float*)d_in[3];   // [E,F,D]
    const float* w2     = (const float*)d_in[4];   // [E,D,F]
    float* out = (float*)d_out;                    // [T,D]

    float* gate_full = (float*)d_ws;                              // T*E*4 = 128 KB
    __bf16* Hbuf = (__bf16*)((char*)d_ws + (size_t)TT * EE * 4);  // T*F*2 = 8 MB

    moe_gate_kernel<<<TT / 256, 256, 0, stream>>>(x, gate_w, gate_full);

    dim3 g1(TT / BM, FF / BN);
    dim3 g2(TT / BM, DD / BN);
    for (int e = 0; e < EE; ++e) {
        moe_gemm1_kernel<<<g1, 256, 0, stream>>>(
            x, w1 + (size_t)e * FF * DD, w3 + (size_t)e * FF * DD, Hbuf);
        moe_gemm2_kernel<<<g2, 256, 0, stream>>>(
            Hbuf, w2 + (size_t)e * DD * FF, gate_full, out, e, e != 0);
    }
}
